// SNNNet_88424786690541
// MI455X (gfx1250) — matrix operand order repaired and checked by reading, not compiled, not run on hardware
//
#include <hip/hip_runtime.h>

// ---------------------------------------------------------------------------
// Fused single-step SNN (snntorch Leaky, zero-init membranes):
//   cur1 = x @ W1^T + b1 ; spk1 = cur1 > 1
//   mem2 = spk1 @ W2^T + b2 ; spk2 = mem2 > 1
//
// Pass 1: convert x -> bf16 rows; repack W1 -> bf16 in WMMA B-fragment order.
// Pass 2: 16384x4096x1024 GEMM via v_wmma_f32_16x16x32_bf16.
//   - 32x1024 bf16 x-tile staged to LDS by the Tensor Data Mover
//     (tensor_load_to_lds + s_wait_tensorcnt), rows padded to 2080B pitch.
//   - Each wave owns a 32x64 output strip: 8 accumulators, each B fragment
//     feeds 2 WMMAs (1 global_load_b128 per WMMA -> half the L2 traffic).
//   - Layer 2 (N=4) fused into the epilogue via LDS float atomics.
// ---------------------------------------------------------------------------

typedef __attribute__((ext_vector_type(16))) __bf16        v16bf;
typedef __attribute__((ext_vector_type(8)))  float         v8f;
typedef __attribute__((ext_vector_type(4)))  unsigned int  v4u;
typedef __attribute__((ext_vector_type(8)))  int           v8i;
typedef __attribute__((ext_vector_type(4)))  int           v4i;

#define B_ROWS   16384
#define N_IN     1024
#define N_HID    4096
#define N_OUT    4
#define BM       32            // batch rows per block (2 M-tiles)
#define NTILES   (N_HID / 16)  // 256 N-tiles
#define KCHUNKS  (N_IN / 32)   // 32 K-chunks of 32
#define NT_WAVE  4             // N-tiles per wave per strip (strip = 32x64)
// TDM pads 4 dwords (16B) after every 256 dwords (1024B):
// x row = 2048B data -> 2080B LDS pitch
#define ROW_PITCH 2080

union AFrag {
    v16bf          v;
    unsigned short u[16];
    uint4          q[2];
};

__device__ __forceinline__ unsigned int f2bf(float f) {
    // fp32 -> bf16, round-to-nearest-even
    unsigned int u = __float_as_uint(f);
    u += 0x7FFFu + ((u >> 16) & 1u);
    return u >> 16;
}

// ---- pass 1a: x (fp32) -> contiguous bf16 rows ----------------------------
__global__ __launch_bounds__(256)
void convert_x_kernel(const float* __restrict__ src, unsigned short* __restrict__ dst)
{
    const size_t n4 = (size_t)B_ROWS * N_IN / 4;
    size_t i = (size_t)blockIdx.x * blockDim.x + threadIdx.x;
    const size_t stride = (size_t)gridDim.x * blockDim.x;
    for (; i < n4; i += stride) {
        float4 v = ((const float4*)src)[i];
        uint2 p;
        p.x = f2bf(v.x) | (f2bf(v.y) << 16);
        p.y = f2bf(v.z) | (f2bf(v.w) << 16);
        ((uint2*)dst)[i] = p;
    }
}

// ---- pass 1b: W1 (fp32 [4096,1024]) -> bf16 in B-fragment order -----------
// layout: [nt (256)][kc (32)][lane (32)][16 bf16]
//   lane l: N = l%16 -> h = nt*16 + l%16 ; K = kc*32 + (l/16)*16 + j
__global__ __launch_bounds__(256)
void repack_w1_kernel(const float* __restrict__ W1, unsigned short* __restrict__ dst)
{
    const int t    = blockIdx.x * 256 + threadIdx.x;   // 0 .. 256*32*32-1
    const int lane = t & 31;
    const int kc   = (t >> 5) & 31;
    const int nt   = t >> 10;
    const int h    = nt * 16 + (lane & 15);
    const int k0   = kc * 32 + (lane >> 4) * 16;

    const float4* s = (const float4*)(W1 + (size_t)h * N_IN + k0);
    float4 a = s[0], b = s[1], c = s[2], d = s[3];
    uint4 o0, o1;
    o0.x = f2bf(a.x) | (f2bf(a.y) << 16);
    o0.y = f2bf(a.z) | (f2bf(a.w) << 16);
    o0.z = f2bf(b.x) | (f2bf(b.y) << 16);
    o0.w = f2bf(b.z) | (f2bf(b.w) << 16);
    o1.x = f2bf(c.x) | (f2bf(c.y) << 16);
    o1.y = f2bf(c.z) | (f2bf(c.w) << 16);
    o1.z = f2bf(d.x) | (f2bf(d.y) << 16);
    o1.w = f2bf(d.z) | (f2bf(d.w) << 16);
    ((uint4*)dst)[(size_t)t * 2 + 0] = o0;
    ((uint4*)dst)[(size_t)t * 2 + 1] = o1;
}

// ---- pass 2: fused WMMA GEMM + LIF ----------------------------------------
__global__ __launch_bounds__(256)
void snn_wmma_kernel(const unsigned short* __restrict__ xbf,
                     const unsigned short* __restrict__ w1p,
                     const float* __restrict__ b1,
                     const float* __restrict__ W2,
                     const float* __restrict__ b2,
                     float* __restrict__ out)
{
    __shared__ __attribute__((aligned(16))) unsigned char xs[BM * ROW_PITCH];
    __shared__ float oacc[BM][N_OUT];

    const int tid  = threadIdx.x;
    const int row0 = blockIdx.x * BM;
    const int wave = tid >> 5;
    const int lane = tid & 31;
    const int l16  = lane & 15;   // A: M row | B/D: N column
    const int hi   = lane >> 4;   // K-half selector

    if (tid < BM * N_OUT)
        oacc[tid / N_OUT][tid % N_OUT] = 0.0f;

    // ---- TDM: stage 32x1024 bf16 x-tile into LDS (padded rows) ------------
    if (wave == 0) {
        unsigned           ldsoff = (unsigned)(uintptr_t)(&xs[0]);
        unsigned long long ga     = (unsigned long long)(uintptr_t)(xbf + (size_t)row0 * N_IN);
        v4u g0;
        g0.x = 1u;                                   // count=1 (valid D#)
        g0.y = ldsoff;                               // lds_addr
        g0.z = (unsigned)(ga & 0xFFFFFFFFu);         // global_addr[31:0]
        g0.w = (unsigned)((ga >> 32) & 0x01FFFFFFu)  // global_addr[56:32]
             | 0x80000000u;                          // type=2 (image)
        v8i g1;
        g1[0] = (int)((1u << 16)      // data_size = 1 -> 2 bytes
                    | (1u << 20)      // pad_enable
                    | (7u << 22)      // pad_interval: 256 dwords (1024B)
                    | (3u << 25));    // pad_amount: 4 dwords (16B)
        g1[1] = (int)(1024u << 16);   // tensor_dim0[15:0] = 1024
        g1[2] = (int)((unsigned)BM << 16); // tensor_dim0 hi=0 | tensor_dim1 = 32
        g1[3] = (int)(1024u << 16);   // tensor_dim1 hi=0 | tile_dim0 = 1024
        g1[4] = BM;                   // tile_dim1 = 32, tile_dim2 = 0
        g1[5] = 1024;                 // tensor_dim0_stride = 1024 elements
        g1[6] = 0;
        g1[7] = 0;
        v4i gz4 = {0, 0, 0, 0};             // groups 2/3 unused (2D tensor)
        v8i gz8 = {0, 0, 0, 0, 0, 0, 0, 0}; // extra group (clang-23 6-arg form)
        __builtin_amdgcn_tensor_load_to_lds(g0, g1, gz4, gz4, gz8, 0);
        __builtin_amdgcn_s_wait_tensorcnt(0);
    }
    __syncthreads();

    const uint4* wq = (const uint4*)w1p;

    // 8 strips per wave; each strip = 32x64 output tile (8 accumulators)
    for (int s = 0; s < NTILES / (8 * NT_WAVE); ++s) {
        const int nt0 = s * 32 + wave * NT_WAVE;

        v8f acc[2 * NT_WAVE];
        #pragma unroll
        for (int i = 0; i < 2 * NT_WAVE; ++i) acc[i] = (v8f){};

        #pragma unroll 2
        for (int kc = 0; kc < KCHUNKS; ++kc) {
            // two A fragments from padded LDS rows (M 0-15 and 16-31)
            const int k0 = kc * 32 + hi * 16;
            const int k1 = k0 + 8;
            const int c0 = 2 * k0 + ((k0 >> 9) << 4);
            const int c1 = 2 * k1 + ((k1 >> 9) << 4);
            AFrag a0, a1;
            a0.q[0] = *(const uint4*)&xs[l16 * ROW_PITCH + c0];
            a0.q[1] = *(const uint4*)&xs[l16 * ROW_PITCH + c1];
            a1.q[0] = *(const uint4*)&xs[(l16 + 16) * ROW_PITCH + c0];
            a1.q[1] = *(const uint4*)&xs[(l16 + 16) * ROW_PITCH + c1];

            // B fragments: contiguous, coalesced 32B/lane; each feeds 2 WMMAs
            #pragma unroll
            for (int t = 0; t < NT_WAVE; ++t) {
                AFrag b;
                size_t base = (((size_t)(nt0 + t) * 32 + kc) * 32 + lane) * 2;
                b.q[0] = wq[base];
                b.q[1] = wq[base + 1];
                acc[2 * t + 0] = __builtin_amdgcn_wmma_f32_16x16x32_bf16(
                    false, a0.v, false, b.v, (short)0, acc[2 * t + 0], false, false);
                acc[2 * t + 1] = __builtin_amdgcn_wmma_f32_16x16x32_bf16(
                    false, a1.v, false, b.v, (short)0, acc[2 * t + 1], false, false);
            }
        }

        // ---- epilogue: threshold + fused layer-2 accumulation -------------
        // D layout: lane's 8 values are rows M = r + 8*hi (+16 for a1 half),
        // column N = l16
        #pragma unroll
        for (int t = 0; t < NT_WAVE; ++t) {
            const int h = (nt0 + t) * 16 + l16;
            const float b1h = b1[h];
            float w2o[N_OUT];
            #pragma unroll
            for (int o = 0; o < N_OUT; ++o)
                w2o[o] = W2[(size_t)o * N_HID + h];

            #pragma unroll
            for (int mh = 0; mh < 2; ++mh) {
                v8f accv = acc[2 * t + mh];
                #pragma unroll
                for (int r = 0; r < 8; ++r) {
                    float cur = accv[r] + b1h;   // mem1 = cur1 (mem_prev == 0)
                    if (cur > 1.0f) {            // spk1 fires
                        int m = r + 8 * hi + 16 * mh;
                        #pragma unroll
                        for (int o = 0; o < N_OUT; ++o)
                            atomicAdd(&oacc[m][o], w2o[o]);
                    }
                }
            }
        }
    }

    __syncthreads();

    // ---- layer-2 epilogue: mem2 / spk2 ------------------------------------
    if (tid < BM * N_OUT) {
        int m = tid / N_OUT, o = tid % N_OUT;
        float mem2 = oacc[m][o] + b2[o];
        size_t idx = (size_t)(row0 + m) * N_OUT + o;
        out[idx] = mem2;                                              // mem2
        out[(size_t)B_ROWS * N_OUT + idx] = (mem2 > 1.0f) ? 1.0f : 0.0f; // spk2
    }
}

extern "C" void kernel_launch(void* const* d_in, const int* in_sizes, int n_in,
                              void* d_out, int out_size, void* d_ws, size_t ws_size,
                              hipStream_t stream) {
    (void)in_sizes; (void)n_in; (void)out_size; (void)ws_size;
    const float* x  = (const float*)d_in[0];
    const float* W1 = (const float*)d_in[1];
    const float* b1 = (const float*)d_in[2];
    const float* W2 = (const float*)d_in[3];
    const float* b2 = (const float*)d_in[4];
    // d_in[5] = beta: multiplied by zero-initialized membranes -> unused.
    float* out = (float*)d_out;

    // workspace layout: [0, 8MB) repacked W1 bf16 ; [8MB, 40MB) x bf16
    unsigned short* w1p = (unsigned short*)d_ws;
    unsigned short* xbf = (unsigned short*)((char*)d_ws + (size_t)N_HID * N_IN * 2);

    repack_w1_kernel<<<dim3(NTILES * KCHUNKS * 32 / 256), dim3(256), 0, stream>>>(W1, w1p);
    convert_x_kernel<<<dim3(1024), dim3(256), 0, stream>>>(x, xbf);
    snn_wmma_kernel<<<dim3(B_ROWS / BM), dim3(256), 0, stream>>>(xbf, w1p, b1, W2, b2, out);
}
